// CausalWanSelfAttention_37443524886952
// MI455X (gfx1250) — compile-verified, matrix operand order checked by reading, not run on hardware
//
#include <hip/hip_runtime.h>
#include <hip/hip_bf16.h>

#define DIM 1536
#define NH 12
#define HD 128
#define MAX_ATTN (15 * 880)                 // 13200
#define LW_PAD (((MAX_ATTN) + 31) & ~31)    // 13216, padded so 32-key chunks stay in-bounds
#define EPS 1e-6f

typedef __attribute__((ext_vector_type(16))) __bf16 v16bf;
typedef __attribute__((ext_vector_type(8)))  float  v8f;

union BFrag {
    v16bf v;
    uint4 q[2];
};

__device__ __forceinline__ v8f vzero8() {
    v8f z;
    for (int e = 0; e < 8; ++e) z[e] = 0.0f;
    return z;
}

__device__ __forceinline__ v8f wmma_bf16(v16bf a, v16bf b, v8f c) {
    // D = A(16x32 bf16) * B(32x16 bf16) + C(16x16 f32)
    return __builtin_amdgcn_wmma_f32_16x16x32_bf16(false, a, false, b, (short)0, c, false, false);
}

// ---------------------------------------------------------------------------
// fp32 -> bf16 elementwise convert
// ---------------------------------------------------------------------------
__global__ void k_f32_to_bf16(const float* __restrict__ src, __bf16* __restrict__ dst, int n) {
    int i = blockIdx.x * blockDim.x + threadIdx.x;
    if (i < n) dst[i] = (__bf16)src[i];
}

// ---------------------------------------------------------------------------
// C[M x N] (f32) = A[M x K] (bf16, row-major) * B[N x K]^T (bf16, row-major) + bias[N]
// One wave (32 threads) computes a 64x64 tile; 16 v_wmma per 32-deep K step.
// ---------------------------------------------------------------------------
__global__ __launch_bounds__(32)
void k_gemm_bias(const __bf16* __restrict__ A, const __bf16* __restrict__ B,
                 const float* __restrict__ bias, float* __restrict__ C,
                 int M, int N, int K) {
    const int lane = threadIdx.x;
    const int l16  = lane & 15;
    const int half = lane >> 4;
    const int n0 = blockIdx.x * 64;
    const int m0 = blockIdx.y * 64;

    v8f acc[4][4];
    for (int i = 0; i < 4; ++i)
        for (int j = 0; j < 4; ++j) acc[i][j] = vzero8();

    for (int k = 0; k < K; k += 32) {
        BFrag a[4], b[4];
#pragma unroll
        for (int i = 0; i < 4; ++i) {
            int row = m0 + i * 16 + l16;
            if (row < M) {
                const __bf16* p = A + (size_t)row * K + k + half * 8;
                a[i].q[0] = *(const uint4*)p;
                a[i].q[1] = *(const uint4*)(p + 16);
                if (k + 64 < K) __builtin_prefetch(p + 64, 0, 1);  // global_prefetch_b8
            } else {
                a[i].q[0] = make_uint4(0, 0, 0, 0);
                a[i].q[1] = make_uint4(0, 0, 0, 0);
            }
        }
#pragma unroll
        for (int j = 0; j < 4; ++j) {
            int col = n0 + j * 16 + l16;
            const __bf16* p = B + (size_t)col * K + k + half * 8;
            b[j].q[0] = *(const uint4*)p;
            b[j].q[1] = *(const uint4*)(p + 16);
        }
#pragma unroll
        for (int i = 0; i < 4; ++i)
#pragma unroll
            for (int j = 0; j < 4; ++j)
                acc[i][j] = wmma_bf16(a[i].v, b[j].v, acc[i][j]);
    }

#pragma unroll
    for (int i = 0; i < 4; ++i) {
#pragma unroll
        for (int j = 0; j < 4; ++j) {
            int col = n0 + j * 16 + l16;
            float bv = bias[col];
#pragma unroll
            for (int e = 0; e < 8; ++e) {
                int row = m0 + i * 16 + e + half * 8;  // C/D layout: vgpr e -> M = e + 8*half
                if (row < M) C[(size_t)row * N + col] = acc[i][j][e] + bv;
            }
        }
    }
}

// ---------------------------------------------------------------------------
// Per-token RMSNorm (over DIM) + RoPE for Q and K; also writes the *new*
// rotated K and plain V into the attention window buffers (bf16).
//   rq    : [NH][S][HD]
//   kwin  : [NH][LW_PAD][HD]
//   vwinT : [NH][HD][LW_PAD]   (transposed so PV B-fragments load contiguously)
// ---------------------------------------------------------------------------
__global__ __launch_bounds__(256)
void k_norm_rope(const float* __restrict__ qf, const float* __restrict__ kf,
                 const float* __restrict__ vf,
                 const float* __restrict__ gq, const float* __restrict__ gk,
                 const float* __restrict__ freqs,
                 const int* __restrict__ grid_sizes, const int* __restrict__ curs,
                 __bf16* __restrict__ rq, __bf16* __restrict__ kwin,
                 __bf16* __restrict__ vwinT, int S) {
    const int t = blockIdx.x;
    const int tid = threadIdx.x;
    const int cs = curs[0];
    const int gh = grid_sizes[1], gw = grid_sizes[2];
    const int frame_seqlen = gh * gw;
    const int start_frame = cs / frame_seqlen;
    const int current_end = cs + S;
    const int win_start = max(0, current_end - MAX_ATTN);
    const int cs_off = cs - win_start;

    const float* qrow = qf + (size_t)t * DIM;
    const float* krow = kf + (size_t)t * DIM;

    float sq = 0.f, sk = 0.f;
    for (int i = tid; i < DIM; i += 256) {
        float a = qrow[i]; sq += a * a;
        float b = krow[i]; sk += b * b;
    }
    for (int off = 16; off > 0; off >>= 1) {
        sq += __shfl_xor(sq, off, 32);
        sk += __shfl_xor(sk, off, 32);
    }
    __shared__ float redq[8], redk[8];
    if ((tid & 31) == 0) { redq[tid >> 5] = sq; redk[tid >> 5] = sk; }
    __syncthreads();
    float sqt = 0.f, skt = 0.f;
    for (int wv = 0; wv < 8; ++wv) { sqt += redq[wv]; skt += redk[wv]; }
    const float qs = rsqrtf(sqt / (float)DIM + EPS);
    const float ks = rsqrtf(skt / (float)DIM + EPS);

    const int hidx = (t / gw) % gh;
    const int widx = t % gw;
    const int fidx = t / frame_seqlen;
    const int C2 = HD / 2;              // 64
    const int s0 = C2 - 2 * (C2 / 3);   // 22
    const int s1 = C2 / 3;              // 21

    for (int p = tid; p < NH * C2; p += 256) {
        int h = p / C2, c = p % C2;
        int row = (c < s0) ? (start_frame + fidx) : ((c < s0 + s1) ? hidx : widx);
        float cv = freqs[((size_t)row * C2 + c) * 2 + 0];
        float sv = freqs[((size_t)row * C2 + c) * 2 + 1];
        int base = h * HD + 2 * c;
        float qre = qrow[base] * qs * gq[base];
        float qim = qrow[base + 1] * qs * gq[base + 1];
        float kre = krow[base] * ks * gk[base];
        float kim = krow[base + 1] * ks * gk[base + 1];
        float qr = qre * cv - qim * sv, qi = qre * sv + qim * cv;
        float kr = kre * cv - kim * sv, ki = kre * sv + kim * cv;
        size_t qo = ((size_t)h * S + t) * HD + 2 * c;
        rq[qo] = (__bf16)qr;
        rq[qo + 1] = (__bf16)qi;
        size_t ko = ((size_t)h * LW_PAD + cs_off + t) * HD + 2 * c;
        kwin[ko] = (__bf16)kr;
        kwin[ko + 1] = (__bf16)ki;
    }
    for (int p = tid; p < NH * HD; p += 256) {
        int h = p / HD, d = p % HD;
        float vv = vf[(size_t)t * DIM + h * HD + d];
        vwinT[((size_t)h * HD + d) * LW_PAD + cs_off + t] = (__bf16)vv;
    }
}

// ---------------------------------------------------------------------------
// Fill window positions [0, cs_off) from the (input, read-only) k/v caches.
// ---------------------------------------------------------------------------
__global__ void k_fill_window(const float* __restrict__ kc, const float* __restrict__ vc,
                              const int* __restrict__ curs,
                              __bf16* __restrict__ kwin, __bf16* __restrict__ vwinT, int S) {
    const int cs = curs[0];
    const int current_end = cs + S;
    const int win_start = max(0, current_end - MAX_ATTN);
    const int cs_off = cs - win_start;
    size_t idx = (size_t)blockIdx.x * blockDim.x + threadIdx.x;
    size_t total = (size_t)cs_off * NH * HD;
    if (idx >= total) return;
    int d = (int)(idx % HD);
    int h = (int)((idx / HD) % NH);
    int j = (int)(idx / ((size_t)HD * NH));
    size_t src = (((size_t)(win_start + j)) * NH + h) * HD + d;
    kwin[((size_t)h * LW_PAD + j) * HD + d] = (__bf16)kc[src];
    vwinT[((size_t)h * HD + d) * LW_PAD + j] = (__bf16)vc[src];
}

// ---------------------------------------------------------------------------
// Split-K flash attention partial pass: 1 wave per (16-query tile, head,
// key-partition). Online softmax within the partition; writes partial
// O (f32 16x128), m, l for the combine pass.
// ---------------------------------------------------------------------------
__global__ __launch_bounds__(32)
void k_attn_part(const __bf16* __restrict__ rq, const __bf16* __restrict__ kwin,
                 const __bf16* __restrict__ vwinT, const int* __restrict__ curs,
                 float* __restrict__ part_o, float* __restrict__ part_m,
                 float* __restrict__ part_l, int S, int QT, int nsplit) {
    const int lane = threadIdx.x;
    const int l16  = lane & 15;
    const int half = lane >> 4;
    const int qt = blockIdx.x;
    const int h  = blockIdx.y;
    const int pp = blockIdx.z;
    const int cs = curs[0];
    const int current_end = cs + S;
    const int win_start = max(0, current_end - MAX_ATTN);
    const int L = current_end - win_start;
    const float scale = 0.08838834764831845f;  // 1/sqrt(128)

    // this partition's key range (chunk is a multiple of 32)
    const int chunk = ((L + nsplit * 32 - 1) / (nsplit * 32)) * 32;
    const int k_begin = pp * chunk;
    const int k_end = min(L, k_begin + chunk);

    // Q fragments: A layout, rows = 16 queries of this tile, K-dim = head dim
    BFrag aq[4];
    {
        int qrow = qt * 16 + l16;
        const __bf16* qb = rq + ((size_t)h * S + min(qrow, S - 1)) * HD;
        bool valid = qrow < S;
#pragma unroll
        for (int dk = 0; dk < 4; ++dk) {
            if (valid) {
                const __bf16* p = qb + dk * 32 + half * 8;
                aq[dk].q[0] = *(const uint4*)p;
                aq[dk].q[1] = *(const uint4*)(p + 16);
            } else {
                aq[dk].q[0] = make_uint4(0, 0, 0, 0);
                aq[dk].q[1] = make_uint4(0, 0, 0, 0);
            }
        }
    }

    v8f o[8];
    for (int j = 0; j < 8; ++j) o[j] = vzero8();
    float mrow[8], lrow[8];
    for (int e = 0; e < 8; ++e) { mrow[e] = -1e30f; lrow[e] = 0.0f; }

    __shared__ __align__(16) __bf16 pls[16][32];

    const __bf16* kbase = kwin + (size_t)h * LW_PAD * HD;
    const __bf16* vbase = vwinT + (size_t)h * HD * LW_PAD;

    for (int kc = k_begin; kc < k_end; kc += 32) {
        v8f s0t = vzero8(), s1t = vzero8();
        const int key0 = kc + l16;
        const int key1 = kc + 16 + l16;
#pragma unroll
        for (int dk = 0; dk < 4; ++dk) {
            BFrag b0, b1;
            if (key0 < L) {
                const __bf16* p0 = kbase + (size_t)key0 * HD + dk * 32 + half * 8;
                b0.q[0] = *(const uint4*)p0;
                b0.q[1] = *(const uint4*)(p0 + 16);
            } else {
                b0.q[0] = make_uint4(0, 0, 0, 0);
                b0.q[1] = make_uint4(0, 0, 0, 0);
            }
            if (key1 < L) {
                const __bf16* p1 = kbase + (size_t)key1 * HD + dk * 32 + half * 8;
                b1.q[0] = *(const uint4*)p1;
                b1.q[1] = *(const uint4*)(p1 + 16);
            } else {
                b1.q[0] = make_uint4(0, 0, 0, 0);
                b1.q[1] = make_uint4(0, 0, 0, 0);
            }
            s0t = wmma_bf16(aq[dk].v, b0.v, s0t);
            s1t = wmma_bf16(aq[dk].v, b1.v, s1t);
        }
        // scale + mask out-of-window keys (column = l16 in both wave halves)
        const bool v0 = key0 < L, v1 = key1 < L;
        float cmax[8];
#pragma unroll
        for (int e = 0; e < 8; ++e) {
            s0t[e] = v0 ? s0t[e] * scale : -1e30f;
            s1t[e] = v1 ? s1t[e] * scale : -1e30f;
            cmax[e] = fmaxf(s0t[e], s1t[e]);
        }
        // per-row max over 16 columns (rows live in vgpr index, same half)
        for (int off = 8; off > 0; off >>= 1)
#pragma unroll
            for (int e = 0; e < 8; ++e) cmax[e] = fmaxf(cmax[e], __shfl_xor(cmax[e], off, 32));
        float alpha[8], psum[8];
#pragma unroll
        for (int e = 0; e < 8; ++e) {
            float mn = fmaxf(mrow[e], cmax[e]);
            alpha[e] = __expf(mrow[e] - mn);
            mrow[e] = mn;
            float p0 = __expf(s0t[e] - mn);
            float p1 = __expf(s1t[e] - mn);
            s0t[e] = p0; s1t[e] = p1;
            psum[e] = p0 + p1;
        }
        for (int off = 8; off > 0; off >>= 1)
#pragma unroll
            for (int e = 0; e < 8; ++e) psum[e] += __shfl_xor(psum[e], off, 32);
#pragma unroll
        for (int e = 0; e < 8; ++e) lrow[e] = lrow[e] * alpha[e] + psum[e];

        // P tile -> LDS (D layout) -> A fragment (16 rows x 32 keys)
#pragma unroll
        for (int e = 0; e < 8; ++e) {
            int row = e + half * 8;
            pls[row][l16] = (__bf16)s0t[e];
            pls[row][l16 + 16] = (__bf16)s1t[e];
        }
        __syncthreads();
        BFrag ap;
        {
            const __bf16* p = &pls[l16][half * 8];
            ap.q[0] = *(const uint4*)p;
            ap.q[1] = *(const uint4*)(p + 16);
        }
        __syncthreads();

        // O = O*alpha + P @ V    (V^T layout -> contiguous B fragments)
#pragma unroll
        for (int dt = 0; dt < 8; ++dt) {
#pragma unroll
            for (int e = 0; e < 8; ++e) o[dt][e] *= alpha[e];
            BFrag bv;
            const __bf16* p = vbase + (size_t)(dt * 16 + l16) * LW_PAD + kc + half * 8;
            bv.q[0] = *(const uint4*)p;
            bv.q[1] = *(const uint4*)(p + 16);
            o[dt] = wmma_bf16(ap.v, bv.v, o[dt]);
        }
    }

    // store partials: base index for (h, qt, pp)
    const size_t pbase = ((size_t)h * QT + qt) * nsplit + pp;
    float* po = part_o + pbase * (16 * HD);
#pragma unroll
    for (int dt = 0; dt < 8; ++dt) {
#pragma unroll
        for (int e = 0; e < 8; ++e) {
            int row = e + half * 8;
            po[(size_t)row * HD + dt * 16 + l16] = o[dt][e];
        }
    }
    if (l16 == 0) {
#pragma unroll
        for (int e = 0; e < 8; ++e) {
            int row = e + half * 8;
            part_m[pbase * 16 + row] = mrow[e];
            part_l[pbase * 16 + row] = lrow[e];
        }
    }
}

// ---------------------------------------------------------------------------
// Combine split-K partials with global log-sum-exp; writes bf16 attention
// output laid out [S][DIM] for the output-projection GEMM.
// Block = 128 threads (one per head-dim column), grid = (QT, NH).
// ---------------------------------------------------------------------------
__global__ __launch_bounds__(128)
void k_attn_combine(const float* __restrict__ part_o, const float* __restrict__ part_m,
                    const float* __restrict__ part_l, __bf16* __restrict__ attn_bf,
                    int S, int QT, int nsplit) {
    const int d = threadIdx.x;
    const int qt = blockIdx.x;
    const int h = blockIdx.y;
    const size_t pbase = ((size_t)h * QT + qt) * nsplit;

    for (int r = 0; r < 16; ++r) {
        int row = qt * 16 + r;
        if (row >= S) break;
        float M = -1e30f;
        for (int p = 0; p < nsplit; ++p)
            M = fmaxf(M, part_m[(pbase + p) * 16 + r]);
        float Lsum = 0.f, acc = 0.f;
        for (int p = 0; p < nsplit; ++p) {
            float w = __expf(part_m[(pbase + p) * 16 + r] - M);
            Lsum += part_l[(pbase + p) * 16 + r] * w;
            acc += part_o[(pbase + p) * (size_t)(16 * HD) + (size_t)r * HD + d] * w;
        }
        attn_bf[(size_t)row * DIM + h * HD + d] = (__bf16)(acc / Lsum);
    }
}

// ---------------------------------------------------------------------------
extern "C" void kernel_launch(void* const* d_in, const int* in_sizes, int n_in,
                              void* d_out, int out_size, void* d_ws, size_t ws_size,
                              hipStream_t stream) {
    const float* x  = (const float*)d_in[0];
    const float* wq = (const float*)d_in[1];
    const float* bq = (const float*)d_in[2];
    const float* wk = (const float*)d_in[3];
    const float* bk = (const float*)d_in[4];
    const float* wv = (const float*)d_in[5];
    const float* bv = (const float*)d_in[6];
    const float* wo = (const float*)d_in[7];
    const float* bo = (const float*)d_in[8];
    const float* gq = (const float*)d_in[9];
    const float* gk = (const float*)d_in[10];
    const float* freqs = (const float*)d_in[11];
    const float* kcache = (const float*)d_in[12];
    const float* vcache = (const float*)d_in[13];
    const int* grid_sizes = (const int*)d_in[14];
    const int* curs = (const int*)d_in[15];

    const int S = in_sizes[0] / DIM;  // 880
    const int QT = (S + 15) / 16;     // 55

    size_t off = 0;
    auto carve = [&](size_t bytes) -> void* {
        void* p = (char*)d_ws + off;
        off += (bytes + 255) & ~(size_t)255;
        return p;
    };
    __bf16* xbf   = (__bf16*)carve((size_t)S * DIM * 2);
    __bf16* wqbf  = (__bf16*)carve((size_t)DIM * DIM * 2);
    __bf16* wkbf  = (__bf16*)carve((size_t)DIM * DIM * 2);
    __bf16* wvbf  = (__bf16*)carve((size_t)DIM * DIM * 2);
    __bf16* wobf  = (__bf16*)carve((size_t)DIM * DIM * 2);
    float*  qf    = (float*)carve((size_t)S * DIM * 4);
    float*  kf    = (float*)carve((size_t)S * DIM * 4);
    float*  vf    = (float*)carve((size_t)S * DIM * 4);
    __bf16* rqbf  = (__bf16*)carve((size_t)NH * S * HD * 2);
    __bf16* kwin  = (__bf16*)carve((size_t)NH * LW_PAD * HD * 2);
    __bf16* vwinT = (__bf16*)carve((size_t)NH * HD * LW_PAD * 2);
    __bf16* attnbf = (__bf16*)carve((size_t)S * DIM * 2);

    // Pick the largest split-K factor whose partial buffers fit the workspace
    // (ws_size is a fixed host value -> deterministic choice across calls).
    int nsplit = 8;
    auto part_bytes = [&](int ns) -> size_t {
        size_t o = (size_t)NH * QT * ns * 16 * HD * 4;   // part_o
        size_t ml = (size_t)NH * QT * ns * 16 * 4;       // part_m / part_l
        return ((o + 255) & ~(size_t)255) + 2 * ((ml + 255) & ~(size_t)255);
    };
    while (nsplit > 1 && off + part_bytes(nsplit) > ws_size) nsplit >>= 1;
    float* part_o = (float*)carve((size_t)NH * QT * nsplit * 16 * HD * 4);
    float* part_m = (float*)carve((size_t)NH * QT * nsplit * 16 * 4);
    float* part_l = (float*)carve((size_t)NH * QT * nsplit * 16 * 4);
    (void)n_in; (void)out_size;

    const int nx = S * DIM, nw = DIM * DIM;
    k_f32_to_bf16<<<(nx + 255) / 256, 256, 0, stream>>>(x, xbf, nx);
    k_f32_to_bf16<<<(nw + 255) / 256, 256, 0, stream>>>(wq, wqbf, nw);
    k_f32_to_bf16<<<(nw + 255) / 256, 256, 0, stream>>>(wk, wkbf, nw);
    k_f32_to_bf16<<<(nw + 255) / 256, 256, 0, stream>>>(wv, wvbf, nw);
    k_f32_to_bf16<<<(nw + 255) / 256, 256, 0, stream>>>(wo, wobf, nw);

    dim3 gg(DIM / 64, (S + 63) / 64);
    k_gemm_bias<<<gg, 32, 0, stream>>>(xbf, wqbf, bq, qf, S, DIM, DIM);
    k_gemm_bias<<<gg, 32, 0, stream>>>(xbf, wkbf, bk, kf, S, DIM, DIM);
    k_gemm_bias<<<gg, 32, 0, stream>>>(xbf, wvbf, bv, vf, S, DIM, DIM);

    k_norm_rope<<<S, 256, 0, stream>>>(qf, kf, vf, gq, gk, freqs, grid_sizes, curs,
                                       rqbf, kwin, vwinT, S);

    size_t fill_total = (size_t)MAX_ATTN * NH * HD;
    k_fill_window<<<(unsigned)((fill_total + 255) / 256), 256, 0, stream>>>(
        kcache, vcache, curs, kwin, vwinT, S);

    dim3 ga(QT, NH, nsplit);
    k_attn_part<<<ga, 32, 0, stream>>>(rqbf, kwin, vwinT, curs,
                                       part_o, part_m, part_l, S, QT, nsplit);

    dim3 gc(QT, NH);
    k_attn_combine<<<gc, 128, 0, stream>>>(part_o, part_m, part_l, attnbf, S, QT, nsplit);

    k_gemm_bias<<<gg, 32, 0, stream>>>(attnbf, wobf, bo, (float*)d_out, S, DIM, DIM);
}
